// RoBERTaGAT_10247791968299
// MI455X (gfx1250) — matrix operand order repaired and checked by reading, not compiled, not run on hardware
//
#include <hip/hip_runtime.h>
#include <hip/hip_bf16.h>
#include <math.h>
#include <stdint.h>

// ---------------- constants ----------------
#define BSZ     128
#define NNODE   200
#define SEQL    128
#define HSD     200
#define SDIM    1024
#define NEDGE   200000
#define NNODES_TOT (BSZ * NNODE)          // 25600
#define ETOT    (NEDGE + NNODES_TOT)      // 225600
#define NHEADS  4
#define DPH     50
#define KLAYERS 5
#define NTAB    624                       // 39 * 4 * 4 distinct edge-feature rows
#define BNSCALE 0.9999950000374997f       // 1/sqrt(1+1e-5)

typedef __attribute__((ext_vector_type(16))) __bf16 v16bf;
typedef __attribute__((ext_vector_type(8)))  float  v8f;
typedef __attribute__((ext_vector_type(4)))  unsigned tdm_g0_t;
typedef __attribute__((ext_vector_type(8)))  int      tdm_g1_t;
typedef __attribute__((ext_vector_type(4)))  int      tdm_g2_t;

// ---------------- small helpers ----------------
__device__ inline float waveReduceSum(float v) {
  for (int off = 16; off > 0; off >>= 1) v += __shfl_xor(v, off, 32);
  return v;
}
__device__ inline unsigned encodeOrd(float f) {
  unsigned u = __float_as_uint(f);
  return (u & 0x80000000u) ? ~u : (u | 0x80000000u);
}
__device__ inline float decodeOrd(unsigned k) {
  unsigned u = (k & 0x80000000u) ? (k & 0x7FFFFFFFu) : ~k;
  return __uint_as_float(u);
}
__device__ inline unsigned pcg_hash(unsigned x) {
  x = x * 747796405u + 2891336453u;
  unsigned w = ((x >> ((x >> 28u) + 4u)) ^ x) * 277803737u;
  return (w >> 22u) ^ w;
}

// ---- Tensor Data Mover: DMA a 2D tile (f32) global -> LDS ----
// dim0/dim1: remaining tensor extent from tile start (elements / rows);
// tile0/tile1: tile shape; OOB reads return zero (engine zero-fills LDS).
__device__ inline void tdm_load_2d_f32(unsigned ldsOff, const float* gaddr,
                                       int dim0, int dim1, int tile0, int tile1,
                                       int strideElems) {
  unsigned long long ga = (unsigned long long)(uintptr_t)gaddr;
  tdm_g0_t g0;
  g0[0] = 1u;                                            // count=1, user mode
  g0[1] = ldsOff;                                        // lds_addr (bytes)
  g0[2] = (unsigned)(ga & 0xFFFFFFFFull);                // global_addr[31:0]
  g0[3] = (unsigned)((ga >> 32) & 0x01FFFFFFull) | (2u << 30);  // addr[56:32] | type=2
  tdm_g1_t g1;
  g1[0] = 0x00020000;                                    // data_size = 2 (4 bytes)
  g1[1] = (int)(((unsigned)dim0 & 0xFFFFu) << 16);       // tensor_dim0[15:0]
  g1[2] = (int)((((unsigned)dim0 >> 16) & 0xFFFFu) | (((unsigned)dim1 & 0xFFFFu) << 16));
  g1[3] = (int)((((unsigned)dim1 >> 16) & 0xFFFFu) | (((unsigned)tile0 & 0xFFFFu) << 16));
  g1[4] = (int)((unsigned)tile1 & 0xFFFFu);              // tile_dim1 (tile_dim2=0)
  g1[5] = strideElems;                                   // tensor_dim0_stride lo32
  g1[6] = 0;
  g1[7] = 0;
  tdm_g2_t gz = {0, 0, 0, 0};
#if __has_include(<hip/amd_detail/amd_gfx1250_TDM.h>)
  tdm_g1_t gz8 = {};
  __builtin_amdgcn_tensor_load_to_lds(g0, g1, gz, gz, gz8, 0);
#else
  __builtin_amdgcn_tensor_load_to_lds(g0, g1, gz, gz, 0);
#endif
}

// ---------------- fill kernels ----------------
__global__ void qag_fill_f32(float* p, float v, long n) {
  long i = (long)blockIdx.x * blockDim.x + threadIdx.x;
  if (i < n) p[i] = v;
}
__global__ void qag_fill_u32(unsigned* p, unsigned v, long n) {
  long i = (long)blockIdx.x * blockDim.x + threadIdx.x;
  if (i < n) p[i] = v;
}

// ---------------- edge prep: src/dst/table-id + out-degree ----------------
__global__ void qag_edge_prep(const int* __restrict__ ei, const int* __restrict__ et,
                              const int* __restrict__ nt,
                              int* srcA, int* dstA, int* tidA, float* cnt) {
  int e = blockIdx.x * blockDim.x + threadIdx.x;
  if (e >= ETOT) return;
  int s, d, t;
  if (e < NEDGE) {
    s = ei[e];
    d = ei[NEDGE + e];
    t = et[e] * 16 + nt[s] * 4 + nt[d];
  } else {
    s = d = e - NEDGE;
    int a = nt[s];
    t = 38 * 16 + a * 4 + a;     // self-loop: one-hot index N_ETYPE
  }
  srcA[e] = s; dstA[e] = d; tidA[e] = t;
  atomicAdd(&cnt[s], 1.0f);
}

// ---------------- edge-encoder layer-1 table (624 x 200) ----------------
__global__ void qag_build_h1tab(const float* __restrict__ w1, const float* __restrict__ b1,
                                const float* __restrict__ g, const float* __restrict__ bb,
                                float* h1) {
  int i = blockIdx.x * blockDim.x + threadIdx.x;
  if (i >= NTAB * HSD) return;
  int t = i / HSD, j = i % HSD;
  int et = t >> 4, a = (t >> 2) & 3, b = t & 3;
  const float* row = w1 + (long)j * 47;
  float v = row[et] + row[39 + a] + row[43 + b] + b1[j];
  v = v * (g[j] * BNSCALE) + bb[j];
  h1[i] = v > 0.f ? v : 0.f;
}

// ---------------- generic WMMA GEMM with TDM tile staging -----------------------
// C[M,Nout] = act(bn((A @ W^T + bias)*alpha)).  A: [M,K] stride lda (f32).
// W: [Nout,K] stride ldw.  Tiles DMA'd to LDS by the Tensor Data Mover (wave 0),
// converted f32->bf16 during fragment packing, v_wmma_f32_16x16x32_bf16 compute.
#define GBM 128
#define GBN 64
#define GBK 32
__global__ __launch_bounds__(256) void qag_gemm_wmma(
    const float* __restrict__ A, int lda, int M, int K,
    const float* __restrict__ W, int ldw,
    float* __restrict__ C, int Nout,
    const float* __restrict__ bias,
    const float* __restrict__ bng, const float* __restrict__ bnb,
    float alpha, int relu) {
  __shared__ float Asf[GBM][GBK];   // 16 KB
  __shared__ float Wsf[GBN][GBK];   //  8 KB
  int tid = threadIdx.x;
  int wave = tid >> 5, lane = tid & 31;
  int rowBase = blockIdx.y * GBM;
  int colBase = blockIdx.x * GBN;
  unsigned asOff = (unsigned)(uintptr_t)&Asf[0][0];
  unsigned wsOff = (unsigned)(uintptr_t)&Wsf[0][0];
  v8f acc[4] = {};
  int kTiles = (K + GBK - 1) / GBK;
  for (int kt = 0; kt < kTiles; ++kt) {
    int k0 = kt * GBK;
    if (tid < 32) {   // one wave drives the TDM; EXEC ignored by tensor ops
      tdm_load_2d_f32(asOff, A + (long)rowBase * lda + k0,
                      K - k0, M - rowBase, GBK, GBM, lda);
      tdm_load_2d_f32(wsOff, W + (long)colBase * ldw + k0,
                      K - k0, Nout - colBase, GBK, GBN, ldw);
      __builtin_amdgcn_s_wait_tensorcnt(0);
    }
    __syncthreads();
    // A fragment: 16x32 bf16, lanes 0-15 hold M=lane K={0..7,16..23}, lanes 16-31 K={8..15,24..31}
    int arow = wave * 16 + (lane & 15);
    int akb = (lane < 16) ? 0 : 8;
    v16bf af;
    #pragma unroll
    for (int j = 0; j < 8; ++j) {
      int kk = ((j < 4) ? 0 : 16) + akb + 2 * (j & 3);
      af[2 * j]     = (__bf16)Asf[arow][kk];
      af[2 * j + 1] = (__bf16)Asf[arow][kk + 1];
    }
    int bcol = lane & 15;
    int bkb = (lane < 16) ? 0 : 8;
    #pragma unroll
    for (int c = 0; c < 4; ++c) {
      v16bf bfv;
      int bn = c * 16 + bcol;
      #pragma unroll
      for (int j = 0; j < 8; ++j) {
        int kk = ((j < 4) ? 0 : 16) + bkb + 2 * (j & 3);
        bfv[2 * j]     = (__bf16)Wsf[bn][kk];
        bfv[2 * j + 1] = (__bf16)Wsf[bn][kk + 1];
      }
      acc[c] = __builtin_amdgcn_wmma_f32_16x16x32_bf16(
          false, af, false, bfv, (short)0, acc[c], false, false);
    }
    __syncthreads();   // all reads done before next TDM overwrites LDS
  }
  // epilogue; C layout: VGPR v -> rows (wave*16 + v) and (+8) for upper half-wave
  #pragma unroll
  for (int c = 0; c < 4; ++c) {
    #pragma unroll
    for (int v = 0; v < 8; ++v) {
      int r = rowBase + wave * 16 + v + ((lane < 16) ? 0 : 8);
      int n = colBase + c * 16 + (lane & 15);
      if (r < M && n < Nout) {
        float t = acc[c][v];
        if (bias) t += bias[n];
        t *= alpha;
        if (bng) t = t * (bng[n] * BNSCALE) + bnb[n];
        if (relu) t = t > 0.f ? t : 0.f;
        C[(long)r * Nout + n] = t;
      }
    }
  }
}

// ---------------- pack xcat = concat(x_cur, nfe) ----------------
__global__ void qag_pack_half(const float* __restrict__ src, float* __restrict__ xcat, int off) {
  long i = (long)blockIdx.x * blockDim.x + threadIdx.x;
  if (i >= (long)NNODES_TOT * HSD) return;
  long n = i / HSD, j = i % HSD;
  xcat[n * (2 * HSD) + off + j] = src[i];
}

// ---------------- pass A: scores + segment max (one wave per edge) ----------------
__global__ __launch_bounds__(256) void qag_scores(
    const float* __restrict__ QxN, const float* __restrict__ KxN,
    const float* __restrict__ KeT,
    const int* __restrict__ srcA, const int* __restrict__ dstA, const int* __restrict__ tidA,
    float* __restrict__ scores, unsigned* __restrict__ segMaxU) {
  int wv = blockIdx.x * 8 + (threadIdx.x >> 5);
  int lane = threadIdx.x & 31;
  if (wv >= ETOT) return;
  int s = srcA[wv], d = dstA[wv], t = tidA[wv];
  const float* q  = QxN + (long)d * HSD;
  const float* kx = KxN + (long)d * HSD;
  const float* ke = KeT + (long)t * HSD;
  if (wv + 1 < ETOT && lane == 0) {
    __builtin_prefetch(QxN + (long)dstA[wv + 1] * HSD, 0, 0);
    __builtin_prefetch(KxN + (long)dstA[wv + 1] * HSD, 0, 0);
  }
  float sc[NHEADS] = {0.f, 0.f, 0.f, 0.f};
  #pragma unroll
  for (int h = 0; h < NHEADS; ++h)
    for (int dd = lane; dd < DPH; dd += 32) {
      int i = h * DPH + dd;
      sc[h] += q[i] * (kx[i] + ke[i]);
    }
  #pragma unroll
  for (int h = 0; h < NHEADS; ++h) sc[h] = waveReduceSum(sc[h]);
  if (lane < NHEADS) {
    float v = sc[lane];
    scores[(long)wv * NHEADS + lane] = v;
    atomicMax(&segMaxU[s * NHEADS + lane], encodeOrd(v));
  }
}

// ---------------- pass B: exp + segment sum ----------------
__global__ void qag_exp(float* __restrict__ scores, const int* __restrict__ srcA,
                        const unsigned* __restrict__ segMaxU, float* __restrict__ segSum) {
  long i = (long)blockIdx.x * blockDim.x + threadIdx.x;
  if (i >= (long)ETOT * NHEADS) return;
  int e = (int)(i >> 2), h = (int)(i & 3);
  int s = srcA[e];
  float m = decodeOrd(segMaxU[s * NHEADS + h]);
  float ex = __expf(scores[i] - m);
  scores[i] = ex;
  atomicAdd(&segSum[s * NHEADS + h], ex);
}

// ---------------- pass C: normalize + scatter-aggregate (one wave per edge) -------
__global__ __launch_bounds__(256) void qag_aggr(
    const float* __restrict__ scores, const float* __restrict__ segSum,
    const float* __restrict__ cnt,
    const float* __restrict__ MxN, const float* __restrict__ MeT,
    const int* __restrict__ srcA, const int* __restrict__ dstA, const int* __restrict__ tidA,
    float* __restrict__ aggr) {
  int wv = blockIdx.x * 8 + (threadIdx.x >> 5);
  int lane = threadIdx.x & 31;
  if (wv >= ETOT) return;
  int s = srcA[wv], d = dstA[wv], t = tidA[wv];
  if (wv + 1 < ETOT && lane == 0)
    __builtin_prefetch(MxN + (long)srcA[wv + 1] * HSD, 0, 0);
  float c = cnt[s];
  float al[NHEADS];
  #pragma unroll
  for (int h = 0; h < NHEADS; ++h)
    al[h] = scores[(long)wv * NHEADS + h] / segSum[s * NHEADS + h] * c;
  const float* mx = MxN + (long)s * HSD;
  const float* me = MeT + (long)t * HSD;
  for (int i = lane; i < HSD; i += 32) {
    int h = i / DPH;
    atomicAdd(&aggr[(long)d * HSD + i], (mx[i] + me[i]) * al[h]);
  }
}

// ---------------- GELU + sensitivity (one wave per node) ----------------
__global__ __launch_bounds__(256) void qag_gelu_sens(
    const float* __restrict__ outB, float* __restrict__ xb, unsigned* __restrict__ sensU) {
  int wv = blockIdx.x * 8 + (threadIdx.x >> 5);
  int lane = threadIdx.x & 31;
  if (wv >= NNODES_TOT) return;
  float ss = 0.f;
  for (int i = lane; i < HSD; i += 32) {
    float v = outB[(long)wv * HSD + i];
    ss += v * v;
    float g = 0.5f * v * (1.f + tanhf(0.7978845608028654f * (v + 0.044715f * v * v * v)));
    xb[(long)wv * HSD + i] = g;
  }
  ss = waveReduceSum(ss);
  if (lane == 0) atomicMax(sensU, encodeOrd(sqrtf(ss)));
}

// ---------------- final rank-1 attention + laplace noise (wave per node) ----------
__global__ __launch_bounds__(256) void qag_final(
    const float* __restrict__ Qb, const float* __restrict__ Kc, const float* __restrict__ Vc,
    const unsigned* __restrict__ sensU, float* __restrict__ xcur, int layer) {
  int wv = blockIdx.x * 8 + (threadIdx.x >> 5);
  int lane = threadIdx.x & 31;
  if (wv >= NNODES_TOT) return;
  int b = wv / NNODE;
  float ss = 0.f, dot = 0.f;
  for (int i = lane; i < HSD; i += 32) {
    float q = Qb[(long)wv * HSD + i];
    ss += q * q;
    dot += q * Kc[(long)b * HSD + i];
  }
  ss  = waveReduceSum(ss);
  dot = waveReduceSum(dot);
  float nrm = fmaxf(sqrtf(ss), 1e-12f);
  float att = dot / nrm * 0.07071067811865475f;   // 1/sqrt(HS)
  float sens = decodeOrd(sensU[0]);               // EP2 == 1
  float tmp = sens * 4.0f;                        // / (EP1/4)
  unsigned h = pcg_hash(0x1234ABCDu ^ (unsigned)(layer * 0x9E3779B9u) ^ ((unsigned)wv * 0x85EBCA6Bu));
  float u01 = (float)(h >> 8) * (1.0f / 16777216.0f);
  u01 = fminf(fmaxf(u01, 1e-7f), 1.f - 1e-7f);
  float p = u01 - 0.5f;
  float lap = -copysignf(logf(1.f - 2.f * fabsf(p)), p);
  float coef = att + lap * tmp;
  for (int i = lane; i < HSD; i += 32)
    xcur[(long)wv * HSD + i] = coef * Vc[(long)b * HSD + i];
}

// ---------------- host side ----------------
static inline void launch_gemm(hipStream_t st, const float* A, int lda, int M, int K,
                               const float* W, int ldw, float* C, int Nout,
                               const float* bias, const float* bng, const float* bnb,
                               float alpha, int relu) {
  dim3 g((Nout + GBN - 1) / GBN, (M + GBM - 1) / GBM);
  qag_gemm_wmma<<<g, 256, 0, st>>>(A, lda, M, K, W, ldw, C, Nout, bias, bng, bnb, alpha, relu);
}
static inline void launch_fill_f(hipStream_t st, float* p, float v, long n) {
  qag_fill_f32<<<(unsigned)((n + 255) / 256), 256, 0, st>>>(p, v, n);
}
static inline void launch_fill_u(hipStream_t st, unsigned* p, unsigned v, long n) {
  qag_fill_u32<<<(unsigned)((n + 255) / 256), 256, 0, st>>>(p, v, n);
}

extern "C" void kernel_launch(void* const* d_in, const int* in_sizes, int n_in,
                              void* d_out, int out_size, void* d_ws, size_t ws_size,
                              hipStream_t stream) {
  (void)in_sizes; (void)n_in; (void)out_size; (void)ws_size;
  // inputs (setup_inputs order)
  const float* hs    = (const float*)d_in[1];
  const float* X0    = (const float*)d_in[2];
  const int*   ei    = (const int*)d_in[3];
  const int*   etyp  = (const int*)d_in[4];
  const int*   ntyp  = (const int*)d_in[5];
  const float* nfe   = (const float*)d_in[6];
  const float* ee_w1 = (const float*)d_in[7];
  const float* ee_b1 = (const float*)d_in[8];
  const float* ee_g  = (const float*)d_in[9];
  const float* ee_bb = (const float*)d_in[10];
  const float* ee_w2 = (const float*)d_in[11];
  const float* ee_b2 = (const float*)d_in[12];
  const float* gk_w  = (const float*)d_in[13];
  const float* gk_b  = (const float*)d_in[14];
  const float* gm_w  = (const float*)d_in[15];
  const float* gm_b  = (const float*)d_in[16];
  const float* gq_w  = (const float*)d_in[17];
  const float* gq_b  = (const float*)d_in[18];
  const float* m1_w  = (const float*)d_in[19];
  const float* m1_b  = (const float*)d_in[20];
  const float* bn_g  = (const float*)d_in[21];
  const float* bn_b  = (const float*)d_in[22];
  const float* m2_w  = (const float*)d_in[23];
  const float* m2_b  = (const float*)d_in[24];
  const float* wq_w  = (const float*)d_in[25];
  const float* wq_b  = (const float*)d_in[26];
  const float* wk_w  = (const float*)d_in[27];
  const float* wk_b  = (const float*)d_in[28];
  const float* wv_w  = (const float*)d_in[29];
  const float* wv_b  = (const float*)d_in[30];
  const float* po_w  = (const float*)d_in[31];
  const float* po_b  = (const float*)d_in[32];
  float* xcur = (float*)d_out;

  // workspace carve-out
  char* base = (char*)d_ws;
  size_t cur = 0;
  auto alloc = [&](size_t bytes) -> char* {
    cur = (cur + 255) & ~(size_t)255;
    char* p = base + cur;
    cur += bytes;
    return p;
  };
  int*      srcA   = (int*)alloc((size_t)ETOT * 4);
  int*      dstA   = (int*)alloc((size_t)ETOT * 4);
  int*      tidA   = (int*)alloc((size_t)ETOT * 4);
  float*    cnt    = (float*)alloc((size_t)NNODES_TOT * 4);
  float*    h1Tab  = (float*)alloc((size_t)NTAB * HSD * 4);
  float*    eTab   = (float*)alloc((size_t)NTAB * HSD * 4);
  float*    KeT    = (float*)alloc((size_t)KLAYERS * NTAB * HSD * 4);
  float*    MeT    = (float*)alloc((size_t)KLAYERS * NTAB * HSD * 4);
  float*    ctx    = (float*)alloc((size_t)BSZ * HSD * 4);
  float*    Kc     = (float*)alloc((size_t)BSZ * HSD * 4);
  float*    Vc     = (float*)alloc((size_t)BSZ * HSD * 4);
  float*    xcat   = (float*)alloc((size_t)NNODES_TOT * 2 * HSD * 4);
  float*    KxN    = (float*)alloc((size_t)NNODES_TOT * HSD * 4);
  float*    MxN    = (float*)alloc((size_t)NNODES_TOT * HSD * 4);
  float*    QxN    = (float*)alloc((size_t)NNODES_TOT * HSD * 4);
  float*    scores = (float*)alloc((size_t)ETOT * NHEADS * 4);
  unsigned* segMaxU= (unsigned*)alloc((size_t)NNODES_TOT * NHEADS * 4);
  float*    segSum = (float*)alloc((size_t)NNODES_TOT * NHEADS * 4);
  float*    aggr   = (float*)alloc((size_t)NNODES_TOT * HSD * 4);
  float*    hbuf   = (float*)alloc((size_t)NNODES_TOT * HSD * 4);
  float*    outB   = (float*)alloc((size_t)NNODES_TOT * HSD * 4);
  float*    xb     = (float*)alloc((size_t)NNODES_TOT * HSD * 4);
  float*    Qb     = (float*)alloc((size_t)NNODES_TOT * HSD * 4);
  unsigned* sensU  = (unsigned*)alloc(4);

  const long NHt = (long)NNODES_TOT * HSD;

  // ---- setup: edge prep + tables + context ----
  launch_fill_f(stream, cnt, 0.f, NNODES_TOT);
  qag_edge_prep<<<(ETOT + 255) / 256, 256, 0, stream>>>(ei, etyp, ntyp, srcA, dstA, tidA, cnt);
  qag_build_h1tab<<<(NTAB * HSD + 255) / 256, 256, 0, stream>>>(ee_w1, ee_b1, ee_g, ee_bb, h1Tab);
  // edge_emb table = h1Tab @ ee_w2^T + ee_b2
  launch_gemm(stream, h1Tab, HSD, NTAB, HSD, ee_w2, HSD, eTab, HSD, ee_b2, nullptr, nullptr, 1.f, 0);
  // per-layer edge-side projection tables (biases folded here)
  for (int l = 0; l < KLAYERS; ++l) {
    launch_gemm(stream, eTab, HSD, NTAB, HSD, gk_w + (size_t)l * HSD * 3 * HSD + 2 * HSD, 3 * HSD,
                KeT + (size_t)l * NTAB * HSD, HSD, gk_b + (size_t)l * HSD, nullptr, nullptr, 1.f, 0);
    launch_gemm(stream, eTab, HSD, NTAB, HSD, gm_w + (size_t)l * HSD * 3 * HSD + 2 * HSD, 3 * HSD,
                MeT + (size_t)l * NTAB * HSD, HSD, gm_b + (size_t)l * HSD, nullptr, nullptr, 1.f, 0);
  }
  // context = hs[:,0,:] @ po_w^T + po_b ; Kc/Vc
  launch_gemm(stream, hs, SEQL * SDIM, BSZ, SDIM, po_w, SDIM, ctx, HSD, po_b, nullptr, nullptr, 1.f, 0);
  launch_gemm(stream, ctx, HSD, BSZ, HSD, wk_w, HSD, Kc, HSD, wk_b, nullptr, nullptr, 1.f, 0);
  launch_gemm(stream, ctx, HSD, BSZ, HSD, wv_w, HSD, Vc, HSD, wv_b, nullptr, nullptr, 1.f, 0);
  // static half of xcat
  qag_pack_half<<<(unsigned)((NHt + 255) / 256), 256, 0, stream>>>(nfe, xcat, HSD);

  // ---- layers ----
  for (int l = 0; l < KLAYERS; ++l) {
    const float* xin = (l == 0) ? X0 : xcur;
    qag_pack_half<<<(unsigned)((NHt + 255) / 256), 256, 0, stream>>>(xin, xcat, 0);
    // node-side projections (biases in tables; qry has bias + 1/sqrt(DPH))
    launch_gemm(stream, xcat, 2 * HSD, NNODES_TOT, 2 * HSD, gk_w + (size_t)l * HSD * 3 * HSD, 3 * HSD,
                KxN, HSD, nullptr, nullptr, nullptr, 1.f, 0);
    launch_gemm(stream, xcat, 2 * HSD, NNODES_TOT, 2 * HSD, gm_w + (size_t)l * HSD * 3 * HSD, 3 * HSD,
                MxN, HSD, nullptr, nullptr, nullptr, 1.f, 0);
    launch_gemm(stream, xcat, 2 * HSD, NNODES_TOT, 2 * HSD, gq_w + (size_t)l * HSD * 2 * HSD, 2 * HSD,
                QxN, HSD, gq_b + (size_t)l * HSD, nullptr, nullptr, 0.1414213562373095f, 0);
    // segment softmax + aggregation
    launch_fill_u(stream, segMaxU, 0u, (long)NNODES_TOT * NHEADS);
    launch_fill_f(stream, segSum, 0.f, (long)NNODES_TOT * NHEADS);
    launch_fill_f(stream, aggr, 0.f, NHt);
    qag_scores<<<(ETOT + 7) / 8, 256, 0, stream>>>(QxN, KxN, KeT + (size_t)l * NTAB * HSD,
                                                   srcA, dstA, tidA, scores, segMaxU);
    qag_exp<<<(unsigned)(((long)ETOT * NHEADS + 255) / 256), 256, 0, stream>>>(scores, srcA, segMaxU, segSum);
    qag_aggr<<<(ETOT + 7) / 8, 256, 0, stream>>>(scores, segSum, cnt, MxN,
                                                 MeT + (size_t)l * NTAB * HSD,
                                                 srcA, dstA, tidA, aggr);
    // MLP: relu(bn(aggr@m1^T + b1)) @ m2^T + b2
    launch_gemm(stream, aggr, HSD, NNODES_TOT, HSD, m1_w + (size_t)l * HSD * HSD, HSD,
                hbuf, HSD, m1_b + (size_t)l * HSD, bn_g + (size_t)l * HSD, bn_b + (size_t)l * HSD, 1.f, 1);
    launch_gemm(stream, hbuf, HSD, NNODES_TOT, HSD, m2_w + (size_t)l * HSD * HSD, HSD,
                outB, HSD, m2_b + (size_t)l * HSD, nullptr, nullptr, 1.f, 0);
    // sensitivity + gelu
    launch_fill_u(stream, sensU, 0x80000000u, 1);
    qag_gelu_sens<<<(NNODES_TOT + 7) / 8, 256, 0, stream>>>(outB, xb, sensU);
    // Q projection, then rank-1 attention with noise
    launch_gemm(stream, xb, HSD, NNODES_TOT, HSD, wq_w, HSD, Qb, HSD, wq_b, nullptr, nullptr, 1.f, 0);
    qag_final<<<(NNODES_TOT + 7) / 8, 256, 0, stream>>>(Qb, Kc, Vc, sensU, xcur, l);
  }
}
